// GCNModel_76957224010204
// MI455X (gfx1250) — compile-verified, matrix operand order checked by reading, not run on hardware
//
#include <hip/hip_runtime.h>

// ---------------------------------------------------------------------------
// Types for CDNA5 WMMA
// ---------------------------------------------------------------------------
typedef __attribute__((ext_vector_type(16))) _Float16 v16h;
typedef __attribute__((ext_vector_type(8)))  float    v8f;

union V16H {
    v16h  v;
    uint4 q[2];   // q[0] = elements 0..7, q[1] = elements 8..15
};

#define N_NODES 100000
#define N_EDGES 1600000
#define IN_C    128
#define HID_C   128
#define OUT_C   64

// ---------------------------------------------------------------------------
// Degree / normalization kernels
// ---------------------------------------------------------------------------
__global__ void init_deg_kernel(float* __restrict__ deg, int n) {
    int i = blockIdx.x * blockDim.x + threadIdx.x;
    if (i < n) deg[i] = 1.0f;   // self-loop weight
}

__global__ void deg_accum_kernel(const long long* __restrict__ dst,
                                 const float* __restrict__ ew,
                                 float* __restrict__ deg, int nE) {
    int e = blockIdx.x * blockDim.x + threadIdx.x;
    if (e < nE) atomicAdd(&deg[dst[e]], ew[e]);
}

__global__ void to_dinv_kernel(float* __restrict__ deg, int n) {
    int i = blockIdx.x * blockDim.x + threadIdx.x;
    if (i < n) {
        float d = deg[i];
        deg[i] = (d > 0.0f) ? rsqrtf(d) : 0.0f;
    }
}

// ---------------------------------------------------------------------------
// Precision-conversion kernels
// ---------------------------------------------------------------------------
__global__ void f32_to_f16_kernel(const float* __restrict__ in,
                                  _Float16* __restrict__ out, size_t n) {
    size_t i = (size_t)blockIdx.x * blockDim.x + threadIdx.x;
    if (i < n) out[i] = (_Float16)in[i];
}

// W: [K, Nc] row-major f32 -> Wt: [Nc, K] f16 (column of W becomes contiguous)
__global__ void transpose_w_kernel(const float* __restrict__ W,
                                   _Float16* __restrict__ Wt, int K, int Nc) {
    int i = blockIdx.x * blockDim.x + threadIdx.x;
    if (i >= K * Nc) return;
    int k = i / Nc, n = i % Nc;
    Wt[(size_t)n * K + k] = (_Float16)W[i];
}

// ---------------------------------------------------------------------------
// WMMA GEMM:  C[M, COUT] = A[M,128] (f16, row major) * B[128, COUT]
//             B supplied transposed as Bt[COUT,128] f16 (Bt[n*128+k] = B[k][n])
// One wave computes one 16x16 output tile: 4 x v_wmma_f32_16x16x32_f16.
// Per-lane ISA layout for 16-bit A (16x32) / B (32x16):
//   lanes 0-15 hold K in {0..7, 16..23}; lanes 16-31 hold K in {8..15, 24..31}
// -> each lane loads two contiguous 16-byte runs of 8 halves.
// ---------------------------------------------------------------------------
template <int COUT>
__global__ __launch_bounds__(256) void gemm_wmma_kernel(
        const _Float16* __restrict__ A,
        const _Float16* __restrict__ Bt,
        float* __restrict__ C) {
    const int lane    = threadIdx.x & 31;
    const int wave    = threadIdx.x >> 5;
    const int rowBase = blockIdx.x << 4;
    const int colBase = wave << 4;          // one column tile per wave
    const int l15     = lane & 15;
    const int khalf   = lane >> 4;          // 0: K lo-half, 1: K hi-half

    const _Float16* ar = A  + (size_t)(rowBase + l15) * 128;
    const _Float16* br = Bt + (size_t)(colBase + l15) * 128;

    v8f acc = {};
#pragma unroll
    for (int kt = 0; kt < 128; kt += 32) {
        const int kb = kt + khalf * 8;
        V16H a, b;
        a.q[0] = *(const uint4*)(ar + kb);       // K = kb .. kb+7
        a.q[1] = *(const uint4*)(ar + kb + 16);  // K = kb+16 .. kb+23
        b.q[0] = *(const uint4*)(br + kb);
        b.q[1] = *(const uint4*)(br + kb + 16);
        acc = __builtin_amdgcn_wmma_f32_16x16x32_f16(
                  false, a.v, false, b.v, (short)0, acc, false, false);
    }

    // C/D layout: VGPR j -> M = khalf*8 + j, N = l15
    float* crow = C + (size_t)(rowBase + khalf * 8) * COUT + colBase + l15;
#pragma unroll
    for (int j = 0; j < 8; ++j)
        crow[(size_t)j * COUT] = acc[j];
}

// ---------------------------------------------------------------------------
// Self-loop contribution + bias:  Out[n,c] = Hs[n,c]*dinv[n]^2 + bias[c]
// ---------------------------------------------------------------------------
__global__ void selfloop_bias_kernel(const float* __restrict__ Hs,
                                     const float* __restrict__ dinv,
                                     const float* __restrict__ bias,
                                     float* __restrict__ Out,
                                     int nNodes, int C) {
    int idx = blockIdx.x * blockDim.x + threadIdx.x;
    if (idx >= nNodes * C) return;
    int n = idx / C, c = idx - n * C;
    float di = dinv[n];
    Out[idx] = Hs[idx] * di * di + bias[c];
}

// ---------------------------------------------------------------------------
// Edge scatter:  Out[dst] += Hs[src] * (dinv[src]*ew*dinv[dst])
// One wave per edge; each lane owns C/32 contiguous channels.
// ---------------------------------------------------------------------------
template <int C>
__global__ __launch_bounds__(256) void edge_scatter_kernel(
        const long long* __restrict__ src,
        const long long* __restrict__ dst,
        const float* __restrict__ ew,
        const float* __restrict__ dinv,
        const float* __restrict__ Hs,
        float* __restrict__ Out, int nE) {
    const int lane = threadIdx.x & 31;
    const int e    = blockIdx.x * (blockDim.x >> 5) + (threadIdx.x >> 5);
    if (e >= nE) return;

    const long long s = src[e];
    const long long d = dst[e];
    const float norm  = dinv[s] * ew[e] * dinv[d];

    const float* hrow = Hs + (size_t)s * C;
    float*       orow = Out + (size_t)d * C;
    __builtin_prefetch(hrow + lane * (C / 32), 0, 1);  // global_prefetch_b8

    const int c0 = lane * (C / 32);
#pragma unroll
    for (int i = 0; i < C / 32; ++i)
        atomicAdd(&orow[c0 + i], hrow[c0 + i] * norm);
}

__global__ void relu_kernel(float* __restrict__ x, size_t n) {
    size_t i = (size_t)blockIdx.x * blockDim.x + threadIdx.x;
    if (i < n) x[i] = fmaxf(x[i], 0.0f);
}

// ---------------------------------------------------------------------------
// Host-side orchestration
// ---------------------------------------------------------------------------
extern "C" void kernel_launch(void* const* d_in, const int* in_sizes, int n_in,
                              void* d_out, int out_size, void* d_ws, size_t ws_size,
                              hipStream_t stream) {
    const float*     x  = (const float*)d_in[0];
    const long long* ei = (const long long*)d_in[1];   // int64 [2, E]
    const float*     ew = (const float*)d_in[2];
    const float*     W1 = (const float*)d_in[3];
    const float*     b1 = (const float*)d_in[4];
    const float*     W2 = (const float*)d_in[5];
    const float*     b2 = (const float*)d_in[6];
    float*           out = (float*)d_out;

    const int nN = in_sizes[0] / IN_C;   // 100000
    const int nE = in_sizes[1] / 2;      // 1600000
    const long long* srcI = ei;
    const long long* dstI = ei + nE;

    // Workspace carve-up (16B-aligned offsets)
    char* ws = (char*)d_ws;
    size_t off = 0;
    float* dinv = (float*)(ws + off); off += ((size_t)nN * 4 + 255) & ~(size_t)255;
    float* hs   = (float*)(ws + off); off += (size_t)nN * HID_C * 4;   // GEMM output
    float* hb   = (float*)(ws + off); off += (size_t)nN * HID_C * 4;   // aggregated
    _Float16* xh  = (_Float16*)(ws + off); off += (size_t)nN * IN_C * 2; // f16 staging
    _Float16* w1t = (_Float16*)(ws + off); off += (size_t)IN_C * HID_C * 2;
    _Float16* w2t = (_Float16*)(ws + off); off += (size_t)HID_C * OUT_C * 2;

    const int TB = 256;
    // ---- degrees / symmetric norm (shared by both layers) ----
    init_deg_kernel<<<(nN + TB - 1) / TB, TB, 0, stream>>>(dinv, nN);
    deg_accum_kernel<<<(nE + TB - 1) / TB, TB, 0, stream>>>(dstI, ew, dinv, nE);
    to_dinv_kernel<<<(nN + TB - 1) / TB, TB, 0, stream>>>(dinv, nN);

    // ---- f16 staging of weights and inputs ----
    transpose_w_kernel<<<(IN_C * HID_C + TB - 1) / TB, TB, 0, stream>>>(W1, w1t, IN_C, HID_C);
    transpose_w_kernel<<<(HID_C * OUT_C + TB - 1) / TB, TB, 0, stream>>>(W2, w2t, HID_C, OUT_C);
    {
        size_t n = (size_t)nN * IN_C;
        f32_to_f16_kernel<<<(unsigned)((n + TB - 1) / TB), TB, 0, stream>>>(x, xh, n);
    }

    const int rowTiles   = nN / 16;                 // 6250 (N divisible by 16)
    const int edgeBlocks = (nE + 7) / 8;            // 8 waves/block, 1 edge/wave

    // ---- Layer 1: hs = x @ W1 (WMMA), aggregate into hb, ReLU ----
    gemm_wmma_kernel<HID_C><<<rowTiles, 256, 0, stream>>>(xh, w1t, hs);
    selfloop_bias_kernel<<<(nN * HID_C + TB - 1) / TB, TB, 0, stream>>>(hs, dinv, b1, hb, nN, HID_C);
    edge_scatter_kernel<HID_C><<<edgeBlocks, 256, 0, stream>>>(srcI, dstI, ew, dinv, hs, hb, nE);
    relu_kernel<<<(unsigned)(((size_t)nN * HID_C + TB - 1) / TB), TB, 0, stream>>>(hb, (size_t)nN * HID_C);

    // ---- Layer 2: h2 = hb @ W2 (WMMA, reuse hs), aggregate into d_out ----
    {
        size_t n = (size_t)nN * HID_C;
        f32_to_f16_kernel<<<(unsigned)((n + TB - 1) / TB), TB, 0, stream>>>(hb, xh, n);
    }
    gemm_wmma_kernel<OUT_C><<<rowTiles, 128, 0, stream>>>(xh, w2t, hs);
    selfloop_bias_kernel<<<(nN * OUT_C + TB - 1) / TB, TB, 0, stream>>>(hs, dinv, b2, out, nN, OUT_C);
    edge_scatter_kernel<OUT_C><<<edgeBlocks, 256, 0, stream>>>(srcI, dstI, ew, dinv, hs, out, nE);
}
